// LegalHeteroGNN_84421877170356
// MI455X (gfx1250) — matrix operand order repaired and checked by reading, not compiled, not run on hardware
//
#include <hip/hip_runtime.h>
#include <hip/hip_bf16.h>
#include <math.h>

// ---------------------------------------------------------------------------
// LegalHeteroGNN for MI455X (gfx1250, wave32).
//  * GEMMs via v_wmma_f32_16x16x32_f16 (f32 accum). Weights are pre-transposed
//    and pre-converted to f16 (Wt[N][K]) so each B fragment is 2x b128 loads;
//    HID-dim activations keep f16 mirrors so A fragments are 2x b128 loads.
//    One wave computes a 64Mx16N strip: B fragment reused across 4 WMMAs.
//  * GAT segment softmax: float-ordered-u32 atomicMax, f32 atomicAdd (L2-resident:
//    node feature tables total ~77MB << 192MB L2).
//  * One wave32 per edge for the 128-wide weighted scatter (alpha computed by
//    lanes 0..3, broadcast via shfl); one wave32 per row for LayerNorm /
//    head dot products; one wave32 per (row, head) for attention scores.
// ---------------------------------------------------------------------------

typedef __attribute__((ext_vector_type(16))) _Float16 v16h;
typedef __attribute__((ext_vector_type(8)))  float    v8f;

#define HID 128
#define NCLAIM_  50000
#define NSTAT_   20000
#define NCASE_   80000

// key(-inf): encode(f) = (u&0x80000000) ? ~u : (u|0x80000000); -inf -> ~0xFF800000
#define AMAX_SENTINEL 0x007FFFFFu

// Fragment layouts (CDNA5 ISA):
//  A (16x32 f16): lane L holds row M=L&15; element i holds
//                 K = (i<8 ? i : i+8) + 8*(L>=16)   (two contiguous 8-runs).
//  B (32x16 f16): lane L holds col N=L&15; element i holds K = (i<8 ? i : i+8) + 8*(L>=16)
//                 (two contiguous 8-runs in Wt[N][K]).
//  C/D (16x16 f32): v8f element r -> M = r + 8*(L>=16), N = L&15.

// ---------------- GEMM core, f16 A operand ----------------------------------
template <int T>
__device__ __forceinline__ void gemm_strip_f16a(
    const _Float16* __restrict__ A, const _Float16* __restrict__ Wt,
    const float* __restrict__ bias, float* __restrict__ C,
    int K, int N, int relu, int m0, int n0, int lane)
{
  const int rc = lane & 15;
  const int kA = (lane >> 4) << 3;     // +8  K offset for lanes 16..31 (A frag)
  const int kB = (lane >> 4) << 3;     // +8  K offset for lanes 16..31 (B frag)
  v8f acc[T];
#pragma unroll
  for (int t = 0; t < T; ++t) acc[t] = (v8f){};

  for (int k0 = 0; k0 < K; k0 += 32) {
    v16h bf;
    const _Float16* bcol = Wt + (size_t)(n0 + rc) * K + k0 + kB;
#pragma unroll
    for (int i = 0; i < 8; ++i) { bf[i] = bcol[i]; bf[i + 8] = bcol[i + 16]; }
#pragma unroll
    for (int t = 0; t < T; ++t) {
      v16h af;
      const _Float16* arow = A + (size_t)(m0 + t * 16 + rc) * K + k0 + kA;
#pragma unroll
      for (int i = 0; i < 8; ++i) { af[i] = arow[i]; af[i + 8] = arow[i + 16]; }
      acc[t] = __builtin_amdgcn_wmma_f32_16x16x32_f16(false, af, false, bf,
                                                      (short)0, acc[t], false, false);
    }
  }
  const int nc = n0 + rc;
  const float bv = bias ? bias[nc] : 0.0f;
#pragma unroll
  for (int t = 0; t < T; ++t) {
    const int mb = m0 + t * 16 + ((lane >> 4) << 3);
#pragma unroll
    for (int r = 0; r < 8; ++r) {
      float v = acc[t][r] + bv;
      if (relu) v = fmaxf(v, 0.0f);
      C[(size_t)(mb + r) * N + nc] = v;
    }
  }
}

// ---------------- GEMM core, f32 A operand (embedding GEMMs) ----------------
template <int T>
__device__ __forceinline__ void gemm_strip_f32a(
    const float* __restrict__ A, const _Float16* __restrict__ Wt,
    const float* __restrict__ bias, float* __restrict__ C,
    int K, int N, int relu, int m0, int n0, int lane)
{
  const int rc = lane & 15;
  const int kA = (lane >> 4) << 3;
  const int kB = (lane >> 4) << 3;
  v8f acc[T];
#pragma unroll
  for (int t = 0; t < T; ++t) acc[t] = (v8f){};

  for (int k0 = 0; k0 < K; k0 += 32) {
    v16h bf;
    const _Float16* bcol = Wt + (size_t)(n0 + rc) * K + k0 + kB;
#pragma unroll
    for (int i = 0; i < 8; ++i) { bf[i] = bcol[i]; bf[i + 8] = bcol[i + 16]; }
#pragma unroll
    for (int t = 0; t < T; ++t) {
      v16h af;
      const float* arow = A + (size_t)(m0 + t * 16 + rc) * K + k0 + kA;
#pragma unroll
      for (int j = 0; j < 8; ++j) {
        const int kb = (j < 4) ? (2 * j) : (16 + 2 * (j - 4));
        const float2 av = *reinterpret_cast<const float2*>(arow + kb);
        af[2 * j]     = (_Float16)av.x;
        af[2 * j + 1] = (_Float16)av.y;
      }
      acc[t] = __builtin_amdgcn_wmma_f32_16x16x32_f16(false, af, false, bf,
                                                      (short)0, acc[t], false, false);
    }
  }
  const int nc = n0 + rc;
  const float bv = bias ? bias[nc] : 0.0f;
#pragma unroll
  for (int t = 0; t < T; ++t) {
    const int mb = m0 + t * 16 + ((lane >> 4) << 3);
#pragma unroll
    for (int r = 0; r < 8; ++r) {
      float v = acc[t][r] + bv;
      if (relu) v = fmaxf(v, 0.0f);
      C[(size_t)(mb + r) * N + nc] = v;
    }
  }
}

__global__ __launch_bounds__(128) void gemm_wmma_f16a_kernel(
    const _Float16* __restrict__ A, const _Float16* __restrict__ Wt,
    const float* __restrict__ bias, float* __restrict__ C,
    int M, int K, int N, int relu)
{
  const int lane = threadIdx.x & 31;
  const int wave = threadIdx.x >> 5;
  const int m0 = blockIdx.x * 64;
  const int n0 = (blockIdx.y * 4 + wave) * 16;
  if (n0 >= N) return;                       // whole wave exits together
  const int rem = (M - m0) >> 4;             // M always a multiple of 16
  const int mt = rem < 4 ? rem : 4;          // wave-uniform
  switch (mt) {
    case 4:  gemm_strip_f16a<4>(A, Wt, bias, C, K, N, relu, m0, n0, lane); break;
    case 3:  gemm_strip_f16a<3>(A, Wt, bias, C, K, N, relu, m0, n0, lane); break;
    case 2:  gemm_strip_f16a<2>(A, Wt, bias, C, K, N, relu, m0, n0, lane); break;
    default: gemm_strip_f16a<1>(A, Wt, bias, C, K, N, relu, m0, n0, lane); break;
  }
}

__global__ __launch_bounds__(128) void gemm_wmma_f32a_kernel(
    const float* __restrict__ A, const _Float16* __restrict__ Wt,
    const float* __restrict__ bias, float* __restrict__ C,
    int M, int K, int N, int relu)
{
  const int lane = threadIdx.x & 31;
  const int wave = threadIdx.x >> 5;
  const int m0 = blockIdx.x * 64;
  const int n0 = (blockIdx.y * 4 + wave) * 16;
  if (n0 >= N) return;
  const int rem = (M - m0) >> 4;
  const int mt = rem < 4 ? rem : 4;
  switch (mt) {
    case 4:  gemm_strip_f32a<4>(A, Wt, bias, C, K, N, relu, m0, n0, lane); break;
    case 3:  gemm_strip_f32a<3>(A, Wt, bias, C, K, N, relu, m0, n0, lane); break;
    case 2:  gemm_strip_f32a<2>(A, Wt, bias, C, K, N, relu, m0, n0, lane); break;
    default: gemm_strip_f32a<1>(A, Wt, bias, C, K, N, relu, m0, n0, lane); break;
  }
}

// ---- weight transpose+convert: Wt[n*K+k] = (f16) W[k*N+n] ------------------
__global__ void wt_convert_kernel(const float* __restrict__ W,
                                  _Float16* __restrict__ Wt, int K, int N)
{
  const int i = blockIdx.x * blockDim.x + threadIdx.x;
  if (i >= K * N) return;
  const int n = i / K, k = i - n * K;
  Wt[i] = (_Float16)W[(size_t)k * N + n];
}

// ---- activation f32 -> f16 mirror ------------------------------------------
__global__ void f16cast_kernel(const float* __restrict__ x,
                               _Float16* __restrict__ y, long long n)
{
  const long long i = (long long)blockIdx.x * blockDim.x + threadIdx.x;
  if (i < n) y[i] = (_Float16)x[i];
}

// ------------- attention scores: a[n,h] = sum_c h[n,h*32+c]*att[h,c] --------
__global__ __launch_bounds__(128) void att_score_kernel(
    const float* __restrict__ h, const float* __restrict__ att,
    float* __restrict__ out, int n)
{
  const int row = blockIdx.x;
  if (row >= n) return;
  const int head = threadIdx.x >> 5, ch = threadIdx.x & 31;
  float v = h[(size_t)row * HID + head * 32 + ch] * att[head * 32 + ch];
#pragma unroll
  for (int o = 16; o > 0; o >>= 1) v += __shfl_xor(v, o, 32);
  if (ch == 0) out[row * 4 + head] = v;
}

// ---------------- fill (u32 pattern; also used for f32 zeroing) -------------
__global__ void fill_u32_kernel(unsigned* __restrict__ p, unsigned v, long long n)
{
  long long i = (long long)blockIdx.x * blockDim.x + threadIdx.x;
  if (i < n) p[i] = v;
}

__device__ __forceinline__ float leaky02(float a) { return a > 0.0f ? a : 0.2f * a; }

// ---------------- edge pass 1: segment max (ordered-u32 atomicMax) ----------
__global__ void edge_max_kernel(const int* __restrict__ src, const int* __restrict__ dst,
                                const float* __restrict__ aS, const float* __restrict__ aD,
                                unsigned* __restrict__ amax, int E)
{
  int e = blockIdx.x * blockDim.x + threadIdx.x;
  if (e >= E) return;
  const int s = src[e], d = dst[e];
#pragma unroll
  for (int h = 0; h < 4; ++h) {
    const float a = leaky02(aS[s * 4 + h] + aD[d * 4 + h]);
    const unsigned u = __float_as_uint(a);
    const unsigned key = (u & 0x80000000u) ? ~u : (u | 0x80000000u);
    atomicMax(&amax[d * 4 + h], key);
  }
}

// decode keys back to floats in place; non-finite (no-edge segments) -> 0
__global__ void amax_finalize_kernel(unsigned* __restrict__ amax, int n)
{
  int i = blockIdx.x * blockDim.x + threadIdx.x;
  if (i >= n) return;
  const unsigned k = amax[i];
  float f = (k & 0x80000000u) ? __uint_as_float(k ^ 0x80000000u)
                              : __uint_as_float(~k);
  if (!__builtin_isfinite(f)) f = 0.0f;
  amax[i] = __float_as_uint(f);
}

// ---------------- edge pass 2: exp-sum into denom ---------------------------
__global__ void edge_sum_kernel(const int* __restrict__ src, const int* __restrict__ dst,
                                const float* __restrict__ aS, const float* __restrict__ aD,
                                const float* __restrict__ amaxf, float* __restrict__ denom,
                                int E)
{
  int e = blockIdx.x * blockDim.x + threadIdx.x;
  if (e >= E) return;
  const int s = src[e], d = dst[e];
#pragma unroll
  for (int h = 0; h < 4; ++h) {
    const float a = leaky02(aS[s * 4 + h] + aD[d * 4 + h]);
    atomicAdd(&denom[d * 4 + h], __expf(a - amaxf[d * 4 + h]));
  }
}

// ---------------- edge pass 3: alpha-weighted scatter (1 wave / edge) -------
__global__ __launch_bounds__(256) void edge_scatter_kernel(
    const int* __restrict__ src, const int* __restrict__ dst,
    const float* __restrict__ aS, const float* __restrict__ aD,
    const float* __restrict__ amaxf, const float* __restrict__ denom,
    const float* __restrict__ hsrc, float* __restrict__ acc, int E)
{
  const int lane = threadIdx.x & 31;
  const int e = blockIdx.x * (blockDim.x >> 5) + (threadIdx.x >> 5);
  if (e >= E) return;
  const int s = src[e], d = dst[e];
  float alpha = 0.0f;
  if (lane < 4) {
    const float a = leaky02(aS[s * 4 + lane] + aD[d * 4 + lane]);
    const float ea = __expf(a - amaxf[d * 4 + lane]);
    alpha = ea / (denom[d * 4 + lane] + 1e-16f);
  }
#pragma unroll
  for (int h = 0; h < 4; ++h) {
    const float al = __shfl(alpha, h, 32);
    const float v = hsrc[(size_t)s * HID + h * 32 + lane] * al;
    atomicAdd(&acc[(size_t)d * HID + h * 32 + lane], v);
  }
}

// --------- LayerNorm + ReLU: x = relu(LN(scale*(acc + b1 + b2)) ) -----------
__global__ __launch_bounds__(256) void ln_relu_kernel(
    const float* __restrict__ acc, const float* __restrict__ b1,
    const float* __restrict__ b2, float scale,
    const float* __restrict__ g, const float* __restrict__ b,
    float* __restrict__ out, int n)
{
  const int lane = threadIdx.x & 31;
  const int row = blockIdx.x * (blockDim.x >> 5) + (threadIdx.x >> 5);
  if (row >= n) return;
  float v[4]; float s = 0.0f;
#pragma unroll
  for (int j = 0; j < 4; ++j) {
    const int ch = j * 32 + lane;
    float t = acc[(size_t)row * HID + ch] + b1[ch] + (b2 ? b2[ch] : 0.0f);
    t *= scale; v[j] = t; s += t;
  }
#pragma unroll
  for (int o = 16; o > 0; o >>= 1) s += __shfl_xor(s, o, 32);
  const float mu = s * (1.0f / 128.0f);
  float var = 0.0f;
#pragma unroll
  for (int j = 0; j < 4; ++j) { const float dd = v[j] - mu; var += dd * dd; }
#pragma unroll
  for (int o = 16; o > 0; o >>= 1) var += __shfl_xor(var, o, 32);
  const float inv = rsqrtf(var * (1.0f / 128.0f) + 1e-5f);
#pragma unroll
  for (int j = 0; j < 4; ++j) {
    const int ch = j * 32 + lane;
    const float y = (v[j] - mu) * inv * g[ch] + b[ch];
    out[(size_t)row * HID + ch] = fmaxf(y, 0.0f);
  }
}

// --------- head output: out[row] = sigmoid(dot(h[row], W2) + b2) ------------
__global__ __launch_bounds__(256) void head_out_kernel(
    const float* __restrict__ h, const float* __restrict__ W2,
    const float* __restrict__ b2, float* __restrict__ out, int n)
{
  const int lane = threadIdx.x & 31;
  const int row = blockIdx.x * (blockDim.x >> 5) + (threadIdx.x >> 5);
  if (row >= n) return;
  float s = 0.0f;
#pragma unroll
  for (int j = 0; j < 4; ++j)
    s += h[(size_t)row * HID + j * 32 + lane] * W2[j * 32 + lane];
#pragma unroll
  for (int o = 16; o > 0; o >>= 1) s += __shfl_xor(s, o, 32);
  if (lane == 0) out[row] = 1.0f / (1.0f + __expf(-(s + b2[0])));
}

// ---------------------------------------------------------------------------
extern "C" void kernel_launch(void* const* d_in, const int* in_sizes, int n_in,
                              void* d_out, int out_size, void* d_ws, size_t ws_size,
                              hipStream_t stream)
{
  (void)n_in; (void)out_size; (void)ws_size;

  // ---- inputs: top-level in setup_inputs() insertion order ----
  const float* x_claim   = (const float*)d_in[0];
  const float* x_statute = (const float*)d_in[1];
  const float* x_case    = (const float*)d_in[2];
  const int* esrc[5] = {(const int*)d_in[3], (const int*)d_in[5], (const int*)d_in[7],
                        (const int*)d_in[9], (const int*)d_in[11]};
  const int* edst[5] = {(const int*)d_in[4], (const int*)d_in[6], (const int*)d_in[8],
                        (const int*)d_in[10], (const int*)d_in[12]};
  int ecnt[5] = {in_sizes[3], in_sizes[5], in_sizes[7], in_sizes[9], in_sizes[11]};

  // ---- params subtree: JAX pytree order (sorted dict keys) starting at 13 ----
  // conv{l}_{e}: {W, att_dst, att_src, bias}  (W < att_dst < att_src < bias)
  // order: conv0_e1..e5, conv1_e1..e5, emb_case{W,b}, emb_claim, emb_statute,
  //        head_coherence{W1,W2,b1,b2}, head_relevance, head_validity,
  //        ln0_case{b,g}, ln0_claim, ln0_statute, ln1_case, ln1_claim, ln1_statute
  auto P = [&](int i) { return (const float*)d_in[i]; };
  auto convBase = [&](int l, int e) { return 13 + (l * 5 + e) * 4; };
  const int EMB_CASE = 53, EMB_CLAIM = 55, EMB_STAT = 57;
  const int HEAD_COH = 59, HEAD_REL = 63, HEAD_VAL = 67;
  const int LN_BASE = 71; // per layer: case(b,g), claim(b,g), statute(b,g)

  // ---- workspace carve-out (float units; all sizes multiples of 32) ----
  float* ws = (float*)d_ws;
  size_t off = 0;
  auto carve = [&](size_t nf) { float* p = ws + off; off += nf; return p; };
  float* xc   = carve((size_t)NCLAIM_ * HID);
  float* xs   = carve((size_t)NSTAT_ * HID);
  float* xk   = carve((size_t)NCASE_ * HID);
  float* accC = carve((size_t)NCLAIM_ * HID);
  float* accS = carve((size_t)NSTAT_ * HID);
  float* accK = carve((size_t)NCASE_ * HID);
  float* hS   = carve((size_t)NCASE_ * HID);   // src projection (also head hidden)
  float* hD   = carve((size_t)NCASE_ * HID);   // dst projection
  float* aS   = carve((size_t)NCASE_ * 4);
  float* aD   = carve((size_t)NCASE_ * 4);
  float* amax = carve((size_t)NCASE_ * 4);     // u32 keys then float maxima
  float* den  = carve((size_t)NCASE_ * 4);
  _Float16* xc16 = (_Float16*)carve((size_t)NCLAIM_ * HID / 2);
  _Float16* xs16 = (_Float16*)carve((size_t)NSTAT_ * HID / 2);
  _Float16* xk16 = (_Float16*)carve((size_t)NCASE_ * HID / 2);
  _Float16* wt   = (_Float16*)carve((size_t)768 * HID / 2);  // max K=768

  auto cdiv = [](long long a, long long b) { return (int)((a + b - 1) / b); };

  auto gemmF32A = [&](const float* A, const float* W, const float* bias, float* C,
                      int M, int K, int relu) {
    wt_convert_kernel<<<cdiv((long long)K * HID, 256), 256, 0, stream>>>(W, wt, K, HID);
    dim3 grid(cdiv(M, 64), HID / 64);
    gemm_wmma_f32a_kernel<<<grid, dim3(128), 0, stream>>>(A, wt, bias, C, M, K, HID, relu);
  };
  auto gemmF16A = [&](const _Float16* A, const float* W, const float* bias, float* C,
                      int M, int K, int relu) {
    wt_convert_kernel<<<cdiv((long long)K * HID, 256), 256, 0, stream>>>(W, wt, K, HID);
    dim3 grid(cdiv(M, 64), HID / 64);
    gemm_wmma_f16a_kernel<<<grid, dim3(128), 0, stream>>>(A, wt, bias, C, M, K, HID, relu);
  };
  auto fill = [&](void* p, unsigned v, long long n) {
    fill_u32_kernel<<<cdiv(n, 256), 256, 0, stream>>>((unsigned*)p, v, n);
  };
  auto cast16 = [&](const float* x, _Float16* y, long long n) {
    f16cast_kernel<<<cdiv(n, 256), 256, 0, stream>>>(x, y, n);
  };

  // runs one GATConv for edge type e of layer l, accumulating into accp
  auto run_conv = [&](int l, int e, const _Float16* xsrc, int ns,
                      const _Float16* xdst, int nd, float* accp) {
    const int base = convBase(l, e);
    const float* W       = P(base + 0);
    const float* att_dst = P(base + 1);
    const float* att_src = P(base + 2);
    gemmF16A(xsrc, W, nullptr, hS, ns, HID, 0);
    const float* hdst = hS;
    if (xsrc != xdst) { gemmF16A(xdst, W, nullptr, hD, nd, HID, 0); hdst = hD; }
    att_score_kernel<<<ns, 128, 0, stream>>>(hS, att_src, aS, ns);
    att_score_kernel<<<nd, 128, 0, stream>>>(hdst, att_dst, aD, nd);
    fill(amax, AMAX_SENTINEL, (long long)nd * 4);
    fill(den, 0u, (long long)nd * 4);
    const int E = ecnt[e];
    edge_max_kernel<<<cdiv(E, 256), 256, 0, stream>>>(esrc[e], edst[e], aS, aD,
                                                      (unsigned*)amax, E);
    amax_finalize_kernel<<<cdiv((long long)nd * 4, 256), 256, 0, stream>>>(
        (unsigned*)amax, nd * 4);
    edge_sum_kernel<<<cdiv(E, 256), 256, 0, stream>>>(esrc[e], edst[e], aS, aD,
                                                      amax, den, E);
    edge_scatter_kernel<<<cdiv(E, 8), 256, 0, stream>>>(esrc[e], edst[e], aS, aD,
                                                        amax, den, hS, accp, E);
  };

  // ---- embeddings (f32 A path; streams the 460MB input once) ----
  gemmF32A(x_claim,   P(EMB_CLAIM), P(EMB_CLAIM + 1), xc, NCLAIM_, 768, 0);
  gemmF32A(x_statute, P(EMB_STAT),  P(EMB_STAT + 1),  xs, NSTAT_,  768, 0);
  gemmF32A(x_case,    P(EMB_CASE),  P(EMB_CASE + 1),  xk, NCASE_,  768, 0);
  cast16(xc, xc16, (long long)NCLAIM_ * HID);
  cast16(xs, xs16, (long long)NSTAT_ * HID);
  cast16(xk, xk16, (long long)NCASE_ * HID);

  // ---- two GNN layers ----
  for (int l = 0; l < 2; ++l) {
    fill(accC, 0u, (long long)NCLAIM_ * HID);
    fill(accS, 0u, (long long)NSTAT_ * HID);
    fill(accK, 0u, (long long)NCASE_ * HID);
    run_conv(l, 0, xk16, NCASE_,  xc16, NCLAIM_, accC);  // e1: case    -> claim
    run_conv(l, 1, xs16, NSTAT_,  xc16, NCLAIM_, accC);  // e2: statute -> claim
    run_conv(l, 2, xc16, NCLAIM_, xk16, NCASE_,  accK);  // e3: claim   -> case
    run_conv(l, 3, xk16, NCASE_,  xk16, NCASE_,  accK);  // e4: case    -> case
    run_conv(l, 4, xc16, NCLAIM_, xs16, NSTAT_,  accS);  // e5: claim   -> statute

    const int lnb = LN_BASE + l * 6;  // case(b,g), claim(b,g), statute(b,g)
    // claim: 0.5*(gat_e1 + gat_e2) -> LN -> relu
    ln_relu_kernel<<<cdiv(NCLAIM_, 8), 256, 0, stream>>>(
        accC, P(convBase(l, 0) + 3), P(convBase(l, 1) + 3), 0.5f,
        P(lnb + 3), P(lnb + 2), xc, NCLAIM_);
    // statute: gat_e5 -> LN -> relu
    ln_relu_kernel<<<cdiv(NSTAT_, 8), 256, 0, stream>>>(
        accS, P(convBase(l, 4) + 3), nullptr, 1.0f,
        P(lnb + 5), P(lnb + 4), xs, NSTAT_);
    // case: 0.5*(gat_e3 + gat_e4) -> LN -> relu
    ln_relu_kernel<<<cdiv(NCASE_, 8), 256, 0, stream>>>(
        accK, P(convBase(l, 2) + 3), P(convBase(l, 3) + 3), 0.5f,
        P(lnb + 1), P(lnb + 0), xk, NCASE_);
    // refresh f16 mirrors for next layer's GEMMs (and the heads after layer 1)
    cast16(xc, xc16, (long long)NCLAIM_ * HID);
    cast16(xs, xs16, (long long)NSTAT_ * HID);
    cast16(xk, xk16, (long long)NCASE_ * HID);
  }

  // ---- output heads on claim nodes; d_out = [validity | relevance | coherence] ----
  float* out = (float*)d_out;
  auto run_head = [&](int base, int outoff) {
    gemmF16A(xc16, P(base + 0), P(base + 2), hS, NCLAIM_, HID, 1);  // relu(x@W1+b1)
    head_out_kernel<<<cdiv(NCLAIM_, 8), 256, 0, stream>>>(
        hS, P(base + 1), P(base + 3), out + outoff, NCLAIM_);
  };
  run_head(HEAD_VAL, 0);
  run_head(HEAD_REL, NCLAIM_);
  run_head(HEAD_COH, 2 * NCLAIM_);
}
